// ResidualBlock_79602923864079
// MI455X (gfx1250) — matrix operand order repaired and checked by reading, not compiled, not run on hardware
//
#include <hip/hip_runtime.h>
#include <hip/hip_bf16.h>

typedef __attribute__((ext_vector_type(16))) _Float16 v16h;
typedef __attribute__((ext_vector_type(8)))  _Float16 v8h;
typedef __attribute__((ext_vector_type(4)))  _Float16 v4h;
typedef __attribute__((ext_vector_type(8)))  float    v8f;

#define NN 131072          // nodes
#define KT 27              // real taps
#define KTP 28             // padded taps (tap 27 zero weights)
#define TAPS_PER_STAGE 4
#define NSTAGE 7           // 7*4 = 28 taps
#define HALVES_PER_TAP 4096        // 2 (c-chunks) * 4 (n-tiles) * 512 halves
#define W_HALVES (KTP * HALVES_PER_TAP)   // 114688 halves per conv
#define ROWS_PER_BLOCK 256         // 8 waves * 2 M-tiles * 16 rows
#define CONV_BLOCKS (NN / ROWS_PER_BLOCK)  // 512

// ---------------------------------------------------------------------------
// Prep: feats f32 -> f16
__global__ void prep_feats(const float* __restrict__ src, _Float16* __restrict__ dst) {
    int i = blockIdx.x * blockDim.x + threadIdx.x;     // over NN*16 float4 groups
    float4 f = ((const float4*)src)[i];
    v4h h;
    h.x = (_Float16)f.x; h.y = (_Float16)f.y; h.z = (_Float16)f.z; h.w = (_Float16)f.w;
    ((v4h*)dst)[i] = h;
}

// Prep: swizzle W[k][ci][co] (f32) into WMMA-B lane layout (f16), zero-pad tap 27.
// B (32x16, f16) layout: lane = {hihalf, n}; per lane 16 halves = K (ci) values
//   halves[j] = W[cc*32 + (lane>>4)*8 + (j<8 ? j : j+8)][t*16 + (lane&15)]
__global__ void prep_weights(const float* __restrict__ W1, const float* __restrict__ W2,
                             _Float16* __restrict__ dst) {
    int e = blockIdx.x * blockDim.x + threadIdx.x;     // 2*W_HALVES threads
    int w  = e / W_HALVES;
    int r  = e % W_HALVES;
    int k  = r >> 12;            // tap
    int r2 = r & 4095;
    int cc = r2 >> 11;           // c-chunk (K 0-31 / 32-63)
    int t  = (r2 >> 9) & 3;      // output n-tile
    int q  = r2 & 511;
    int lane = q >> 4;
    int j    = q & 15;
    int ci = cc * 32 + (lane >> 4) * 8 + ((j < 8) ? j : (j + 8));
    int co = t * 16 + (lane & 15);
    float v = 0.f;
    if (k < KT) {
        const float* W = (w == 0) ? W1 : W2;
        v = W[(k * 64 + ci) * 64 + co];
    }
    dst[e] = (_Float16)v;
}

// A 16x32 f16 fragment: lane<16 holds K 0-7 + 16-23, lane>=16 holds K 8-15 + 24-31
__device__ __forceinline__ v16h load_a_frag(const _Float16* ap, int cc) {
    v8h lo = *(const v8h*)(ap + cc * 32);
    v8h hi = *(const v8h*)(ap + cc * 32 + 16);
    return __builtin_shufflevector(lo, hi, 0,1,2,3,4,5,6,7,8,9,10,11,12,13,14,15);
}

// ---------------------------------------------------------------------------
// Gather-GEMM: per wave, TWO 16-node M-tiles x 64 out-channels; each LDS B tile
// feeds two back-to-back WMMAs (1 ds_load_b128 : 1 wmma).
// APPLY_BN: apply (x*scale+shift, relu) in f16 to gathered A rows (conv2).
// OUT16: write f32 accumulators as f16 (conv1 output feeds next gather).
template<int APPLY_BN, int OUT16>
__global__ void __launch_bounds__(256) conv_kernel(
    const _Float16* __restrict__ src,      // [NN,64] f16
    const int*      __restrict__ nidx,     // [27,NN]
    const _Float16* __restrict__ wswz,     // [W_HALVES] swizzled weights
    const _Float16* __restrict__ bnscale,  // [64] f16 (APPLY_BN only)
    const _Float16* __restrict__ bnshift,  // [64]
    _Float16* __restrict__ out16,          // [NN,64] f16 (OUT16)
    float*    __restrict__ out32,          // [NN,64] f32 (!OUT16)
    float*    __restrict__ partial)        // [gridDim][128]: sum[64], sumsq[64]
{
    __shared__ __attribute__((aligned(32))) _Float16 lb[TAPS_PER_STAGE * HALVES_PER_TAP]; // 32 KB

    const int tid    = threadIdx.x;
    const int lane   = tid & 31;
    const int wave   = tid >> 5;
    const int mlane  = lane & 15;
    const int hihalf = lane >> 4;
    const int m0     = blockIdx.x * ROWS_PER_BLOCK + wave * 32;  // 2 M-tiles per wave
    const int row_m  = m0 + mlane;         // tile-0 node for this lane (A M = mlane)
    const int aoff   = hihalf * 8;         // A K-offset per lane half

    // BN affine vectors matching per-lane A channel positions (per c-chunk)
    v16h s0, s1, b0, b1;
    if (APPLY_BN) {
#pragma unroll
        for (int j = 0; j < 16; ++j) {
            int cj = aoff + ((j < 8) ? j : (j + 8));   // halves[8..15] are K+16
            s0[j] = bnscale[cj];      b0[j] = bnshift[cj];
            s1[j] = bnscale[32 + cj]; b1[j] = bnshift[32 + cj];
        }
    }

    v8f acc0[4] = {};   // tile 0: rows m0..m0+15
    v8f acc1[4] = {};   // tile 1: rows m0+16..m0+31

    for (int stage = 0; stage < NSTAGE; ++stage) {
        // stage 32 KB of swizzled weights into LDS (straight copy, layout matches)
        const uint4* gsrc = (const uint4*)(wswz + stage * (TAPS_PER_STAGE * HALVES_PER_TAP));
        uint4* ldst = (uint4*)lb;
#pragma unroll
        for (int i = 0; i < 8; ++i) ldst[tid + i * 256] = gsrc[tid + i * 256];
        __syncthreads();

        for (int kk = 0; kk < TAPS_PER_STAGE; ++kk) {
            const int k = stage * TAPS_PER_STAGE + kk;
            if (k < KT) {                       // wave-uniform
                const int row0 = nidx[k * NN + row_m];
                const int row1 = nidx[k * NN + row_m + 16];
                const _Float16* ap0 = src + row0 * 64 + aoff;
                const _Float16* ap1 = src + row1 * 64 + aoff;
#pragma unroll
                for (int cc = 0; cc < 2; ++cc) {
                    v16h a0 = load_a_frag(ap0, cc);
                    v16h a1 = load_a_frag(ap1, cc);
                    if (APPLY_BN) {
                        const v16h sv = cc ? s1 : s0;
                        const v16h bv = cc ? b1 : b0;
#pragma unroll
                        for (int j = 0; j < 16; ++j) {
                            _Float16 x0 = a0[j] * sv[j] + bv[j];
                            _Float16 x1 = a1[j] * sv[j] + bv[j];
                            a0[j] = (x0 > (_Float16)0.f) ? x0 : (_Float16)0.f;
                            a1[j] = (x1 > (_Float16)0.f) ? x1 : (_Float16)0.f;
                        }
                    }
                    const _Float16* bb = lb + (kk * 2 + cc) * 2048 + lane * 16;
#pragma unroll
                    for (int t = 0; t < 4; ++t) {
                        v16h b = *(const v16h*)(bb + t * 512);
                        acc0[t] = __builtin_amdgcn_wmma_f32_16x16x32_f16(
                                    false, a0, false, b, (short)0, acc0[t], false, false);
                        acc1[t] = __builtin_amdgcn_wmma_f32_16x16x32_f16(
                                    false, a1, false, b, (short)0, acc1[t], false, false);
                    }
                }
            }
        }
        __syncthreads();
    }

    // ---- store D + deterministic BN partial reduction (reuse LDS) ----
    float* redS = (float*)lb;            // [16 halfwave-slots][4 t][16 n]
    float* redQ = redS + 1024;
#pragma unroll
    for (int t = 0; t < 4; ++t) {
        const int c = t * 16 + mlane;    // output channel held by this lane
        float s = 0.f, q = 0.f;
#pragma unroll
        for (int tile = 0; tile < 2; ++tile) {
            const v8f& A = tile ? acc1[t] : acc0[t];
#pragma unroll
            for (int r = 0; r < 8; ++r) {  // D rows: lane<16 -> M 0-7, lane>=16 -> M 8-15
                float v = A[r];
                s += v; q += v * v;
                const int m = m0 + tile * 16 + hihalf * 8 + r;
                if (OUT16) out16[m * 64 + c] = (_Float16)v;
                else       out32[m * 64 + c] = v;
            }
        }
        redS[((hihalf * 8 + wave) * 4 + t) * 16 + mlane] = s;
        redQ[((hihalf * 8 + wave) * 4 + t) * 16 + mlane] = q;
    }
    __syncthreads();
    if (tid < 64) {
        const int t = tid >> 4, l = tid & 15;
        float s = 0.f, q = 0.f;
#pragma unroll
        for (int h = 0; h < 16; ++h) {
            s += redS[(h * 4 + t) * 16 + l];
            q += redQ[(h * 4 + t) * 16 + l];
        }
        partial[blockIdx.x * 128 + tid]      = s;
        partial[blockIdx.x * 128 + 64 + tid] = q;
    }
}

// ---------------------------------------------------------------------------
// Fixed-order BN finalize: fold gamma/beta into per-channel scale/shift.
__global__ void bn_finalize(const float* __restrict__ partial,
                            const float* __restrict__ gamma,
                            const float* __restrict__ beta,
                            float* __restrict__ sc32, float* __restrict__ sh32,
                            _Float16* __restrict__ sc16, _Float16* __restrict__ sh16) {
    int c = threadIdx.x;
    if (c >= 64) return;
    float s = 0.f, q = 0.f;
    for (int b = 0; b < CONV_BLOCKS; ++b) {
        s += partial[b * 128 + c];
        q += partial[b * 128 + 64 + c];
    }
    float mu  = s * (1.0f / NN);
    float var = q * (1.0f / NN) - mu * mu;
    float inv = rsqrtf(var + 1e-5f);
    float sc  = gamma[c] * inv;
    float sh  = beta[c] - mu * sc;
    sc32[c] = sc; sh32[c] = sh;
    sc16[c] = (_Float16)sc; sh16[c] = (_Float16)sh;
}

// Epilogue: out = relu(h2*scale2 + shift2 + feats)
__global__ void final_kernel(const float* __restrict__ h2,
                             const float* __restrict__ feats,
                             const float* __restrict__ sc, const float* __restrict__ sh,
                             float* __restrict__ out) {
    int i = blockIdx.x * blockDim.x + threadIdx.x;   // over NN*16 float4 groups
    int c = (i & 15) * 4;
    float4 h = ((const float4*)h2)[i];
    float4 f = ((const float4*)feats)[i];
    float4 o;
    o.x = fmaxf(h.x * sc[c + 0] + sh[c + 0] + f.x, 0.f);
    o.y = fmaxf(h.y * sc[c + 1] + sh[c + 1] + f.y, 0.f);
    o.z = fmaxf(h.z * sc[c + 2] + sh[c + 2] + f.z, 0.f);
    o.w = fmaxf(h.w * sc[c + 3] + sh[c + 3] + f.w, 0.f);
    ((float4*)out)[i] = o;
}

// ---------------------------------------------------------------------------
extern "C" void kernel_launch(void* const* d_in, const int* in_sizes, int n_in,
                              void* d_out, int out_size, void* d_ws, size_t ws_size,
                              hipStream_t stream) {
    const float* feats = (const float*)d_in[0];
    const int*   nidx  = (const int*)d_in[1];
    const float* W1    = (const float*)d_in[2];
    const float* g1    = (const float*)d_in[3];
    const float* b1    = (const float*)d_in[4];
    const float* W2    = (const float*)d_in[5];
    const float* g2    = (const float*)d_in[6];
    const float* b2    = (const float*)d_in[7];
    float* out = (float*)d_out;

    char* ws = (char*)d_ws;
    size_t off = 0;
    auto alloc = [&](size_t bytes) -> char* {
        char* p = ws + off;
        off = (off + bytes + 255) & ~(size_t)255;
        return p;
    };
    _Float16* featsH = (_Float16*)alloc((size_t)NN * 64 * 2);
    _Float16* wswz   = (_Float16*)alloc((size_t)2 * W_HALVES * 2);
    _Float16* h1h    = (_Float16*)alloc((size_t)NN * 64 * 2);
    float*    h2     = (float*)   alloc((size_t)NN * 64 * 4);
    float*    part1  = (float*)   alloc((size_t)CONV_BLOCKS * 128 * 4);
    float*    part2  = (float*)   alloc((size_t)CONV_BLOCKS * 128 * 4);
    float*    sc1_32 = (float*)   alloc(256);
    float*    sh1_32 = (float*)   alloc(256);
    float*    sc2_32 = (float*)   alloc(256);
    float*    sh2_32 = (float*)   alloc(256);
    _Float16* sc1_16 = (_Float16*)alloc(128);
    _Float16* sh1_16 = (_Float16*)alloc(128);
    _Float16* sc2_16 = (_Float16*)alloc(128);
    _Float16* sh2_16 = (_Float16*)alloc(128);

    prep_feats<<<NN * 16 / 256, 256, 0, stream>>>(feats, featsH);
    prep_weights<<<2 * W_HALVES / 256, 256, 0, stream>>>(W1, W2, wswz);

    conv_kernel<0, 1><<<CONV_BLOCKS, 256, 0, stream>>>(
        featsH, nidx, wswz, nullptr, nullptr, h1h, nullptr, part1);
    bn_finalize<<<1, 64, 0, stream>>>(part1, g1, b1, sc1_32, sh1_32, sc1_16, sh1_16);

    conv_kernel<1, 0><<<CONV_BLOCKS, 256, 0, stream>>>(
        h1h, nidx, wswz + W_HALVES, sc1_16, sh1_16, nullptr, h2, part2);
    bn_finalize<<<1, 64, 0, stream>>>(part2, g2, b2, sc2_32, sh2_32, sc2_16, sh2_16);

    final_kernel<<<NN * 16 / 256, 256, 0, stream>>>(h2, feats, sc2_32, sh2_32, out);
}
